// GRU_76828374990971
// MI455X (gfx1250) — matrix operand order repaired and checked by reading, not compiled, not run on hardware
//
#include <hip/hip_runtime.h>

// ---------------------------------------------------------------------------
// Two-kernel fused-over-time GRU stack for gfx1250 (MI455X).
//   Layer 0: torch nn.GRUCell  (r,z,n order), K = [x(64) | h(128)] = 192
//   Layer 1: custom GRU        K = [x1(128) | h1 or r*h1 (128)] = 256
// Each wave32 owns a 16-row batch tile for all T=24 steps; h state lives in
// f32 WMMA accumulators. Weights are bf16 in LDS in row-major [out][in]
// (the K-contiguous 16x16x32 bf16 B layout); all fragment addressing is a
// single per-wave base pointer + compile-time offsets (folds into ds_load
// immediates, avoiding address-register pressure / spills).
// CDNA5 data movers (guarded by __has_builtin):
//   - kernel1 stages its x1 tile with the Tensor Data Mover (2D tile + LDS
//     padding to interleave into the 256-wide A tile)
//   - kernel0 streams seq1 out with global_store_async_from_lds_b128
// ---------------------------------------------------------------------------

typedef __attribute__((ext_vector_type(16))) __bf16 v16bf;
typedef __attribute__((ext_vector_type(8)))  float  v8f;
typedef __attribute__((ext_vector_type(4)))  unsigned int v4u;
typedef __attribute__((ext_vector_type(8)))  int v8i_;
typedef __attribute__((ext_vector_type(4)))  int v4i_;

#define AS1 __attribute__((address_space(1)))
#define AS3 __attribute__((address_space(3)))

#define BN_  20800   // 64 * 325
#define T_   24
#define F_   64
#define H_   128
#define MT_  (BN_ / 16)   // 1300 row tiles
#define WPB_ 8            // waves per block

#if __has_builtin(__builtin_amdgcn_tensor_load_to_lds)
#define HAVE_TDM 1
#else
#define HAVE_TDM 0
#endif
#if __has_builtin(__builtin_amdgcn_global_load_async_to_lds_b128)
#define HAVE_ASY_LD 1
#else
#define HAVE_ASY_LD 0
#endif
#if __has_builtin(__builtin_amdgcn_global_store_async_from_lds_b128)
#define HAVE_ASY_ST 1
#else
#define HAVE_ASY_ST 0
#endif

static __device__ __forceinline__ void wait_dscnt0() {
#if __has_builtin(__builtin_amdgcn_s_wait_dscnt)
  __builtin_amdgcn_s_wait_dscnt(0);
#else
  asm volatile("s_wait_dscnt 0x0" ::: "memory");
#endif
}
static __device__ __forceinline__ void wait_async0() {
#if __has_builtin(__builtin_amdgcn_s_wait_asynccnt)
  __builtin_amdgcn_s_wait_asynccnt(0);
#else
  asm volatile("s_wait_asynccnt 0x0" ::: "memory");
#endif
}
static __device__ __forceinline__ void wait_tensor0() {
#if __has_builtin(__builtin_amdgcn_s_wait_tensorcnt)
  __builtin_amdgcn_s_wait_tensorcnt(0);
#else
  asm volatile("s_wait_tensorcnt 0x0" ::: "memory");
#endif
}

static __device__ __forceinline__ unsigned int pk_bf16(float a, float b) {
  unsigned short ua = __builtin_bit_cast(unsigned short, (__bf16)a);
  unsigned short ub = __builtin_bit_cast(unsigned short, (__bf16)b);
  return (unsigned int)ua | ((unsigned int)ub << 16);
}
static __device__ __forceinline__ unsigned short bf16b(float a) {
  return __builtin_bit_cast(unsigned short, (__bf16)a);
}
static __device__ __forceinline__ float sigm(float x) {
  return 1.0f / (1.0f + __builtin_amdgcn_exp2f(-1.44269504088896f * x));
}
static __device__ __forceinline__ float tanh_f(float x) {
  float e = __builtin_amdgcn_exp2f(2.88539008177793f * (-x));
  return (1.0f - e) / (1.0f + e);
}
static __device__ __forceinline__ v8f zero_v8() {
  v8f z = {0.f, 0.f, 0.f, 0.f, 0.f, 0.f, 0.f, 0.f};
  return z;
}

union Frag { uint4 q[2]; v16bf v; };

// A fragment (16x32 bf16): lane = row (+K half), 32B contiguous along K
static __device__ __forceinline__ v16bf ldA(const unsigned short* p) {
  Frag f;
  f.q[0] = *reinterpret_cast<const uint4*>(p);
  f.q[1] = *reinterpret_cast<const uint4*>(p + 8);
  return f.v;
}
// B fragment (32x16 bf16): lane = col (+K half), 32B contiguous along K
static __device__ __forceinline__ v16bf ldB(const unsigned short* p) {
  Frag f;
  f.q[0] = *reinterpret_cast<const uint4*>(p);
  f.q[1] = *reinterpret_cast<const uint4*>(p + 8);
  return f.v;
}

#define WMMA_BF16(A, B, C) \
  __builtin_amdgcn_wmma_f32_16x16x32_bf16(false, (A), false, (B), (short)0, (C), false, false)

// ---------------------------------------------------------------------------
// Kernel 0: torch GRUCell over T steps.  LDS: W 384x192 bf16 (144K) + 8 A-tiles
// ---------------------------------------------------------------------------
__global__ __launch_bounds__(256) void gru_layer0(
    const float* __restrict__ x, const float* __restrict__ w_ih,
    const float* __restrict__ w_hh, const float* __restrict__ b_ih,
    const float* __restrict__ b_hh, unsigned short* __restrict__ seq1) {
  __shared__ __align__(16) unsigned short sW[384 * 192];        // [n][k], k=[x|h]
  __shared__ __align__(16) unsigned short sA[WPB_][16 * 192];   // per-wave act tile

  const int tid = threadIdx.x;
  // Stage weights (f32 -> bf16), layout [n][0..63]=w_ih row, [n][64..191]=w_hh row
  for (int i = tid; i < 384 * 32; i += 256) {           // w_ih: 384x64, pairs
    int n = i >> 5, k2 = (i & 31) * 2;
    const float* s = w_ih + n * 64 + k2;
    *reinterpret_cast<unsigned int*>(&sW[n * 192 + k2]) = pk_bf16(s[0], s[1]);
  }
  for (int i = tid; i < 384 * 64; i += 256) {           // w_hh: 384x128, pairs
    int n = i >> 6, k2 = (i & 63) * 2;
    const float* s = w_hh + n * 128 + k2;
    *reinterpret_cast<unsigned int*>(&sW[n * 192 + 64 + k2]) = pk_bf16(s[0], s[1]);
  }
  __syncthreads();

  const int wave  = tid >> 5;
  const int lane  = tid & 31;
  const int khalf = lane >> 4;
  const int l16   = lane & 15;
  const int mt    = blockIdx.x * WPB_ + wave;
  if (mt >= MT_) return;                       // uniform per wave; no barriers below
  const int m0 = mt * 16;
  unsigned short* aw = sA[wave];

  // zero h region of the A tile (cols 64..191)
  for (int r = 0; r < 16; ++r) {
    uint2 z; z.x = 0u; z.y = 0u;
    *reinterpret_cast<uint2*>(&aw[r * 192 + 64 + lane * 4]) = z;
  }

  // per-lane bias registers (lane owns C-layout column l16 of each 16-col block)
  float brz[16], bin8[8], bhn8[8];
#pragma unroll
  for (int j = 0; j < 16; ++j) brz[j] = b_ih[j * 16 + l16] + b_hh[j * 16 + l16];
#pragma unroll
  for (int j = 0; j < 8; ++j) {
    bin8[j] = b_ih[256 + j * 16 + l16];
    bhn8[j] = b_hh[256 + j * 16 + l16];
  }

  v8f hst[8];
#pragma unroll
  for (int j = 0; j < 8; ++j) hst[j] = zero_v8();

  const int xr = lane >> 1;                 // staging row (2 lanes per row)
  const int xc = (lane & 1) * 32;           // f32 col base for x staging
  const float* xrow = x + ((size_t)(m0 + xr) * T_) * F_ + xc;
  unsigned short* srow = seq1 + ((size_t)(m0 + xr) * T_) * H_ + (lane & 1) * 64;

  // Per-wave fragment base pointers: everything else is a constant offset.
  const unsigned short* wbase = &sW[l16 * 192 + khalf * 16];   // B frags
  const unsigned short* abase = &aw[l16 * 192 + khalf * 16];   // A frags

  for (int t = 0; t < T_; ++t) {
    // ---- stage x_t tile: 16x64 f32 -> bf16 into A cols 0..63
    const float* xp = xrow + (size_t)t * F_;
    if (t + 1 < T_) __builtin_prefetch(xp + F_, 0, 0);   // next step's x row
#pragma unroll
    for (int i = 0; i < 8; ++i) {
      float4 v = *reinterpret_cast<const float4*>(xp + i * 4);
      uint2 p; p.x = pk_bf16(v.x, v.y); p.y = pk_bf16(v.z, v.w);
      *reinterpret_cast<uint2*>(&aw[xr * 192 + xc + i * 4]) = p;
    }
    __builtin_amdgcn_wave_barrier();   // LDS ops are in-order within a wave

    v8f accc[16], acci[8], acch[8];
#pragma unroll
    for (int j = 0; j < 16; ++j) accc[j] = zero_v8();
#pragma unroll
    for (int j = 0; j < 8; ++j) { acci[j] = zero_v8(); acch[j] = zero_v8(); }

    // ---- gates: [16x192] x [192x384]; r,z fused over K; n-gate i/h kept apart
#pragma unroll
    for (int kb = 0; kb < 6; ++kb) {
      v16bf a = ldA(abase + kb * 32);
#pragma unroll
      for (int nb = 0; nb < 16; ++nb)
        accc[nb] = WMMA_BF16(a, ldB(wbase + nb * 3072 + kb * 32), accc[nb]);
#pragma unroll
      for (int nb = 0; nb < 8; ++nb) {
        v16bf b = ldB(wbase + 49152 + nb * 3072 + kb * 32);   // rows 256..383
        if (kb < 2) acci[nb] = WMMA_BF16(a, b, acci[nb]);     // x part (K<64)
        else        acch[nb] = WMMA_BF16(a, b, acch[nb]);     // h part (K>=64)
      }
    }

#if HAVE_ASY_ST
    wait_async0();   // previous step's seq1 async store must be done reading LDS h
#endif
    // ---- elementwise GRU update on C-layout; restage h' into A cols 64..191
#pragma unroll
    for (int j = 0; j < 8; ++j) {
#pragma unroll
      for (int v = 0; v < 8; ++v) {
        float r  = sigm(accc[j][v] + brz[j]);
        float z  = sigm(accc[8 + j][v] + brz[8 + j]);
        float nn = tanh_f(acci[j][v] + bin8[j] + r * (acch[j][v] + bhn8[j]));
        float h  = (1.0f - z) * nn + z * hst[j][v];
        hst[j][v] = h;
        aw[(v + 8 * khalf) * 192 + 64 + j * 16 + l16] = bf16b(h);
      }
    }
    __builtin_amdgcn_wave_barrier();

    // ---- seq1 write (bf16) straight out of the LDS h region
    const unsigned short* hs = &aw[xr * 192 + 64 + (lane & 1) * 64];
    unsigned short* dst = srow + (size_t)t * H_;
#if HAVE_ASY_ST
    wait_dscnt0();   // h restage committed to LDS before async engine reads it
#pragma unroll
    for (int i = 0; i < 8; ++i)
      __builtin_amdgcn_global_store_async_from_lds_b128(
          (AS1 v4i_*)(dst + i * 8), (AS3 v4i_*)(hs + i * 8), 0, 0);
#else
#pragma unroll
    for (int i = 0; i < 8; ++i)
      *reinterpret_cast<uint4*>(dst + i * 8) =
          *reinterpret_cast<const uint4*>(hs + i * 8);
#endif
  }
}

// ---------------------------------------------------------------------------
// Kernel 1: custom GRUCell over T steps.  LDS: gate_w 128K + out_w 64K + tiles
// ---------------------------------------------------------------------------
__global__ __launch_bounds__(256) void gru_layer1(
    const unsigned short* __restrict__ seq1, const float* __restrict__ gate_w,
    const float* __restrict__ gate_b, const float* __restrict__ out_w,
    const float* __restrict__ out_b, float* __restrict__ out) {
  __shared__ __align__(16) unsigned short sG[256 * 256];       // gate_w [n][k]
  __shared__ __align__(16) unsigned short sO[128 * 256];       // out_w  [n][k]
  __shared__ __align__(16) unsigned short sA[WPB_][16 * 256];  // per-wave tiles

  const int tid = threadIdx.x;
  for (int i = tid; i < 256 * 128; i += 256) {     // gate_w pairs (flat copy)
    int idx = i * 2;
    const float* s = gate_w + idx;
    *reinterpret_cast<unsigned int*>(&sG[idx]) = pk_bf16(s[0], s[1]);
  }
  for (int i = tid; i < 128 * 128; i += 256) {     // out_w pairs
    int idx = i * 2;
    const float* s = out_w + idx;
    *reinterpret_cast<unsigned int*>(&sO[idx]) = pk_bf16(s[0], s[1]);
  }
  __syncthreads();

  const int wave  = tid >> 5;
  const int lane  = tid & 31;
  const int khalf = lane >> 4;
  const int l16   = lane & 15;
  const int mt    = blockIdx.x * WPB_ + wave;
  if (mt >= MT_) return;
  const int m0 = mt * 16;
  unsigned short* aw = sA[wave];

  for (int r = 0; r < 16; ++r) {                  // zero h region (cols 128..255)
    uint2 z; z.x = 0u; z.y = 0u;
    *reinterpret_cast<uint2*>(&aw[r * 256 + 128 + lane * 4]) = z;
  }

  float bg[16], bo8[8];
#pragma unroll
  for (int j = 0; j < 16; ++j) bg[j] = gate_b[j * 16 + l16];
#pragma unroll
  for (int j = 0; j < 8; ++j) bo8[j] = out_b[j * 16 + l16];

  v8f hst[8];
#pragma unroll
  for (int j = 0; j < 8; ++j) hst[j] = zero_v8();

  const int xr = lane >> 1;
  const int xc = (lane & 1) * 64;                 // bf16 col base
  const unsigned short* srow = seq1 + ((size_t)(m0 + xr) * T_) * H_ + xc;

  // Per-wave fragment base pointers.
  const unsigned short* gbase = &sG[l16 * 256 + khalf * 16];
  const unsigned short* obase = &sO[l16 * 256 + khalf * 16];
  const unsigned short* abase = &aw[l16 * 256 + khalf * 16];

#if HAVE_TDM
  // Wave-uniform LDS byte offset of this wave's A tile (for the D#).
  unsigned lds_off = __builtin_amdgcn_readfirstlane(
      (unsigned)(unsigned long long)(AS3 void*)aw);
#endif

  for (int t = 0; t < T_; ++t) {
    // ---- stage x1_t (bf16, 16x128, row pitch T*H) into A cols 0..127
    wait_dscnt0();   // all reads of the x region from the previous step done
#if HAVE_TDM
    {
      // TDM 2D tile: tile 128x16 bf16, tensor row stride T*H elements; LDS
      // padding (64 dwords every 64 dwords) interleaves rows into the
      // 256-element-stride A tile without touching the h half.
      unsigned long long ga =
          (unsigned long long)(const void*)(seq1 + ((size_t)m0 * T_ + t) * H_);
      unsigned galo = __builtin_amdgcn_readfirstlane((unsigned)ga);
      unsigned gahi =
          __builtin_amdgcn_readfirstlane((unsigned)(ga >> 32)) & 0x01ffffffu;
      v4u g0 = {1u /*count=1, load, user*/, lds_off, galo, gahi | (2u << 30)};
      v8i_ g1 = {(int)0x7F510000u,   // data_size=2B, pad_en, interval=2^6 dw, amount=64 dw
                 (int)(128u << 16),  // tensor_dim0 = 128 (low 16 bits at bit 48)
                 (int)(16u << 16),   // tensor_dim1 = 16
                 (int)(128u << 16),  // tile_dim0 = 128
                 16,                 // tile_dim1 = 16 (tile_dim2 = 0)
                 (int)(T_ * H_),     // tensor_dim0_stride = 3072 elements
                 0, 0};
      v4i_ gz = {0, 0, 0, 0};
#if defined(__clang_major__) && __clang_major__ >= 23
      v8i_ gz8 = {0, 0, 0, 0, 0, 0, 0, 0};
      __builtin_amdgcn_tensor_load_to_lds(g0, g1, gz, gz, gz8, 0);
#else
      __builtin_amdgcn_tensor_load_to_lds(g0, g1, gz, gz, 0);
#endif
      wait_tensor0();
    }
#elif HAVE_ASY_LD
    {
      const unsigned short* sp = srow + (size_t)t * H_;
#pragma unroll
      for (int i = 0; i < 8; ++i)
        __builtin_amdgcn_global_load_async_to_lds_b128(
            (AS1 v4i_*)(sp + i * 8),
            (AS3 v4i_*)(aw + xr * 256 + xc + i * 8), 0, 0);
      wait_async0();
    }
#else
    {
      const unsigned short* sp = srow + (size_t)t * H_;
#pragma unroll
      for (int i = 0; i < 8; ++i)
        *reinterpret_cast<uint4*>(&aw[xr * 256 + xc + i * 8]) =
            *reinterpret_cast<const uint4*>(sp + i * 8);
      __builtin_amdgcn_wave_barrier();
    }
#endif

    // ---- gates = [x1 | h1] x gate_w^T  (K=256, N=256)
    v8f accg[16];
#pragma unroll
    for (int j = 0; j < 16; ++j) accg[j] = zero_v8();
#pragma unroll
    for (int kb = 0; kb < 8; ++kb) {
      v16bf a = ldA(abase + kb * 32);
#pragma unroll
      for (int nb = 0; nb < 16; ++nb)
        accg[nb] = WMMA_BF16(a, ldB(gbase + nb * 4096 + kb * 32), accg[nb]);
    }

    // ---- r*h -> overwrite A h-region for the candidate matmul
#pragma unroll
    for (int j = 0; j < 8; ++j) {
#pragma unroll
      for (int v = 0; v < 8; ++v) {
        float r = sigm(accg[j][v] + bg[j]);
        aw[(v + 8 * khalf) * 256 + 128 + j * 16 + l16] = bf16b(r * hst[j][v]);
      }
    }
    __builtin_amdgcn_wave_barrier();

    // ---- h_cand = [x1 | r*h1] x out_w^T  (K=256, N=128)
    v8f acco[8];
#pragma unroll
    for (int j = 0; j < 8; ++j) acco[j] = zero_v8();
#pragma unroll
    for (int kb = 0; kb < 8; ++kb) {
      v16bf a = ldA(abase + kb * 32);
#pragma unroll
      for (int nb = 0; nb < 8; ++nb)
        acco[nb] = WMMA_BF16(a, ldB(obase + nb * 4096 + kb * 32), acco[nb]);
    }

    // ---- h' = z*h + (1-z)*tanh(.); write f32 output, restage h' as bf16
#pragma unroll
    for (int j = 0; j < 8; ++j) {
#pragma unroll
      for (int v = 0; v < 8; ++v) {
        float z  = sigm(accg[8 + j][v] + bg[8 + j]);
        float hc = tanh_f(acco[j][v] + bo8[j]);
        float h  = z * hst[j][v] + (1.0f - z) * hc;
        hst[j][v] = h;
        int row = v + 8 * khalf;
        out[((size_t)(m0 + row) * T_ + t) * H_ + j * 16 + l16] = h;
        aw[row * 256 + 128 + j * 16 + l16] = bf16b(h);
      }
    }
  }
}

// ---------------------------------------------------------------------------
extern "C" void kernel_launch(void* const* d_in, const int* in_sizes, int n_in,
                              void* d_out, int out_size, void* d_ws, size_t ws_size,
                              hipStream_t stream) {
  (void)in_sizes; (void)n_in; (void)out_size; (void)ws_size;
  const float* x    = (const float*)d_in[0];
  const float* w_ih = (const float*)d_in[1];
  const float* w_hh = (const float*)d_in[2];
  const float* b_ih = (const float*)d_in[3];
  const float* b_hh = (const float*)d_in[4];
  const float* gw   = (const float*)d_in[5];
  const float* gb   = (const float*)d_in[6];
  const float* ow   = (const float*)d_in[7];
  const float* ob   = (const float*)d_in[8];

  unsigned short* seq1 = (unsigned short*)d_ws;   // bf16 [BN][T][H], ~128 MB
  float* out = (float*)d_out;

  dim3 block(32 * WPB_);
  dim3 grid((MT_ + WPB_ - 1) / WPB_);             // 163 blocks of 8 waves
  gru_layer0<<<grid, block, 0, stream>>>(x, w_ih, w_hh, b_ih, b_hh, seq1);
  gru_layer1<<<grid, block, 0, stream>>>(seq1, gw, gb, ow, ob, out);
}
